// FaceClassifier_38190849196539
// MI455X (gfx1250) — matrix operand order repaired and checked by reading, not compiled, not run on hardware
//
#include <hip/hip_runtime.h>

#define HIDDEN   128
#define KNN_K    20
#define K1PAD    160   // 9 + 128 = 137 inputs, padded to 5 K-tiles of 32
#define SPB      4     // segments (triangles) per block
#define EPB      80    // edges per block = SPB * KNN_K = 5 M-tiles of 16
#define LTHREADS 160   // 5 waves (wave32), one 16-edge M-tile per wave

// fused fp16 weight image per layer: W1[128][160] | W2[128][128]
#define W1HALFS  (HIDDEN * K1PAD)            // 20480
#define W2HALFS  (HIDDEN * HIDDEN)           // 16384
#define SW_BYTES ((W1HALFS + W2HALFS) * 2)   // 73728 bytes
#define WTOT8    (SW_BYTES / 8)              // 9216 8-byte elements (TDM data_size=8)

// dynamic LDS layout (all offsets 32B aligned)
#define OFF_W1   0
#define OFF_W2   (W1HALFS * 2)               // 40960
#define OFF_HIN  SW_BYTES                    // 73728
#define OFF_H1   (OFF_HIN + EPB * K1PAD * 2) // 99328
#define OFF_ACC  (OFF_H1 + EPB * HIDDEN * 2) // 119808
#define OFF_TGT  (OFF_ACC + SPB * HIDDEN * 4)// 121856
#define SMEM_TOTAL (OFF_TGT + EPB * 4)       // 122176

typedef __attribute__((ext_vector_type(16))) _Float16 v16h;
typedef __attribute__((ext_vector_type(8)))  _Float16 v8h;
typedef __attribute__((ext_vector_type(8)))  float    v8f;
typedef __attribute__((ext_vector_type(4)))  unsigned int v4u;
typedef __attribute__((ext_vector_type(8)))  int      v8i;
typedef __attribute__((ext_vector_type(4)))  int      v4i;

union FragU { v16h v; v8h h[2]; };

// ---------------------------------------------------------------------------
// 1. Per-triangle geometry: t_min(3), t_max(3), bary(3) packed as geo[t][9]
// ---------------------------------------------------------------------------
__global__ void geometry_kernel(const float* __restrict__ pts,
                                const int*   __restrict__ tri,
                                float* __restrict__ geo, int T)
{
    int t = blockIdx.x * blockDim.x + threadIdx.x;
    if (t >= T) return;
    int ia = tri[t * 3 + 0], ib = tri[t * 3 + 1], ic = tri[t * 3 + 2];
    #pragma unroll
    for (int c = 0; c < 3; ++c) {
        float a = pts[ia * 3 + c];
        float b = pts[ib * 3 + c];
        float d = pts[ic * 3 + c];
        float e1 = a - b;          // e_ij
        float e2 = a - d;          // e_ik
        float e3 = b - d;          // e_jk
        geo[t * 9 + c]     = fminf(e1, fminf(e2, e3));    // t_min
        geo[t * 9 + 3 + c] = fmaxf(e1, fmaxf(e2, e3));    // t_max
        geo[t * 9 + 6 + c] = (a + b + d) * (1.0f / 3.0f); // bary
    }
}

// ---------------------------------------------------------------------------
// 2. Brute-force KNN (k=20, excluding self). One row per thread, candidate
//    barycenters tiled through LDS, per-thread top-21 insertion list in LDS.
//    Neighbor order is irrelevant downstream (segment_sum is order-invariant).
// ---------------------------------------------------------------------------
__global__ __launch_bounds__(256) void knn_kernel(const float* __restrict__ geo,
                                                  int* __restrict__ nbr, int T)
{
    __shared__ float s_b[256 * 3];
    __shared__ float s_d[21 * 256];
    __shared__ int   s_i[21 * 256];

    const int tid = threadIdx.x;
    const int i   = blockIdx.x * 256 + tid;
    const bool active = (i < T);

    float bx = 0.f, by = 0.f, bz = 0.f;
    if (active) {
        bx = geo[i * 9 + 6];
        by = geo[i * 9 + 7];
        bz = geo[i * 9 + 8];
    }
    #pragma unroll
    for (int p = 0; p < 21; ++p) {
        s_d[p * 256 + tid] = 3.4e38f;
        s_i[p * 256 + tid] = 0;
    }

    for (int j0 = 0; j0 < T; j0 += 256) {
        __syncthreads();
        int j = j0 + tid;
        if (j < T) {
            s_b[tid * 3 + 0] = geo[j * 9 + 6];
            s_b[tid * 3 + 1] = geo[j * 9 + 7];
            s_b[tid * 3 + 2] = geo[j * 9 + 8];
        }
        __syncthreads();
        int jmax = (T - j0 < 256) ? (T - j0) : 256;
        for (int jj = 0; jj < jmax; ++jj) {
            float dx = bx - s_b[jj * 3 + 0];
            float dy = by - s_b[jj * 3 + 1];
            float dz = bz - s_b[jj * 3 + 2];
            float d2 = dx * dx + dy * dy + dz * dz;
            if (active && d2 < s_d[20 * 256 + tid]) {
                int p = 20;
                while (p > 0 && s_d[(p - 1) * 256 + tid] > d2) {
                    s_d[p * 256 + tid] = s_d[(p - 1) * 256 + tid];
                    s_i[p * 256 + tid] = s_i[(p - 1) * 256 + tid];
                    --p;
                }
                s_d[p * 256 + tid] = d2;
                s_i[p * 256 + tid] = j0 + jj;
            }
        }
    }
    if (active) {
        #pragma unroll
        for (int k = 0; k < KNN_K; ++k)          // slot 0 is self (d2 == 0)
            nbr[i * KNN_K + k] = s_i[(k + 1) * 256 + tid];
    }
}

// ---------------------------------------------------------------------------
// 3. fp32 -> fp16 weight conversion with zero K-padding (row-major [n][k])
// ---------------------------------------------------------------------------
__global__ void convert_pad_kernel(const float* __restrict__ src,
                                   _Float16* __restrict__ dst,
                                   int rows, int cin, int cout)
{
    int idx = blockIdx.x * blockDim.x + threadIdx.x;
    if (idx >= rows * cout) return;
    int r = idx / cout, c = idx - r * cout;
    dst[idx] = (_Float16)(c < cin ? src[r * cin + c] : 0.0f);
}

// ---------------------------------------------------------------------------
// 4. feats0 = broadcast(init_probs) over 128 channels
// ---------------------------------------------------------------------------
__global__ void init_feats_kernel(const float* __restrict__ p,
                                  float* __restrict__ x, int T)
{
    int idx = blockIdx.x * blockDim.x + threadIdx.x;
    if (idx < T * HIDDEN) x[idx] = p[idx >> 7];
}

// ---------------------------------------------------------------------------
// 5. Fused TriConv layer.
//    TDM stages the fused fp16 W1|W2 image (73,728 B) into LDS (one
//    tensor_load_to_lds per block, TENSORcnt-tracked, overlapped with the
//    edge-feature gather/build) -> WMMA GEMM1 (+bias, ReLU) -> WMMA GEMM2
//    (+bias) -> in-block segment sum via LDS ds_add_f32.
//    Block = 4 segments = 80 edges; 5 waves, one 16-row M-tile each.
// ---------------------------------------------------------------------------
__global__ __launch_bounds__(LTHREADS) void triconv_kernel(
    const float* __restrict__ geo, const int* __restrict__ nbr,
    const float* __restrict__ x_in,
    const _Float16* __restrict__ wts,                               // W1|W2 fp16
    const float* __restrict__ b1, const float* __restrict__ b2,
    float* __restrict__ x_out, int T)
{
    extern __shared__ char smem[];
    _Float16* s_w1  = (_Float16*)(smem + OFF_W1);
    _Float16* s_w2  = (_Float16*)(smem + OFF_W2);
    _Float16* s_hin = (_Float16*)(smem + OFF_HIN);
    _Float16* s_h1  = (_Float16*)(smem + OFF_H1);
    float*    s_acc = (float*)   (smem + OFF_ACC);
    int*      s_tgt = (int*)     (smem + OFF_TGT);

    const int tid  = threadIdx.x;
    const int wave = tid >> 5;
    const int lane = tid & 31;
    const int seg0 = blockIdx.x * SPB;

    // --- wave 0: kick off TDM copy of the weight image into LDS ------------
    if (wave == 0) {
        unsigned long long ga = (unsigned long long)wts;
        unsigned ldsa = (unsigned)(unsigned long long)(void*)smem; // LDS byte addr
        // D# group0: count=1 | lds_addr | global_addr[56:0] | type=2
        v4u g0;
        g0.x = 1u;                                   // count=1 (valid, load)
        g0.y = ldsa;
        g0.z = (unsigned)ga;
        g0.w = (unsigned)((ga >> 32) & 0x1FFFFFFu) | (2u << 30);
        // D# group1: data_size=8B; 1 x 9216 tile; tensor_dim0 = stride0 = 9216
        v8i g1;
        g1[0] = (int)(3u << 16);                     // wg_mask=0, data_size=3 (8B)
        g1[1] = (int)((WTOT8 & 0xFFFFu) << 16);      // tensor_dim0[15:0]
        g1[2] = (int)((WTOT8 >> 16) | (1u << 16));   // tensor_dim0[31:16] | tensor_dim1=1
        g1[3] = (int)((unsigned)WTOT8 << 16);        // tile_dim0 = 9216
        g1[4] = 1;                                   // tile_dim1 = 1, tile_dim2 = 0
        g1[5] = (int)WTOT8;                          // tensor_dim0_stride[31:0]
        g1[6] = 0;                                   // stride0[47:32] | stride1[15:0]
        g1[7] = 0;                                   // stride1[47:16]
        v4i z4 = (v4i){0, 0, 0, 0};                  // groups 2/3 unused (2D tile)
        v8i z8 = (v8i){0, 0, 0, 0, 0, 0, 0, 0};
        __builtin_amdgcn_tensor_load_to_lds(g0, g1, z4, z4, z8, 0);
    }

    // --- gather neighbor ids for this block's 80 edges ---------------------
    if (tid < EPB) {
        int src = seg0 + tid / KNN_K;
        int v = 0;
        if (src < T) v = nbr[src * KNN_K + (tid % KNN_K)];
        s_tgt[tid] = v;
    }
    for (int i = tid; i < SPB * HIDDEN; i += LTHREADS) s_acc[i] = 0.0f;
    __syncthreads();

    // --- build edge features h_in[e][0:9]=geo diff, [9:137]=x diff, pad 0 --
    for (int idx = tid; idx < EPB * K1PAD; idx += LTHREADS) {
        int e = idx / K1PAD;
        int c = idx - e * K1PAD;
        int src = seg0 + e / KNN_K;
        if (src >= T) src = T - 1;
        int tgt = s_tgt[e];
        float v = 0.0f;
        if (c < 9)        v = geo[src * 9 + c] - geo[tgt * 9 + c];
        else if (c < 137) v = x_in[src * HIDDEN + (c - 9)] - x_in[tgt * HIDDEN + (c - 9)];
        s_hin[idx] = (_Float16)v;
    }
    if (wave == 0) __builtin_amdgcn_s_wait_tensorcnt(0);  // weights landed
    __syncthreads();

    const int nl = lane & 15;   // A: row M; B/C/D: column N
    const int g  = lane >> 4;

    // --- GEMM1: A(16x160) x W1^T, bias, ReLU -> s_h1 fp16 ------------------
    v16h a1[5];
    #pragma unroll
    for (int kt = 0; kt < 5; ++kt) {
        const _Float16* rp = &s_hin[(wave * 16 + nl) * K1PAD + kt * 32 + 16 * g];
        FragU u;
        u.h[0] = *(const v8h*)(rp);        // K = kt*32 + 16g .. +7
        u.h[1] = *(const v8h*)(rp + 8);    // K = kt*32 + 16g + 8 .. +15
        a1[kt] = u.v;
    }

    #pragma unroll
    for (int nt = 0; nt < 8; ++nt) {
        const float binit = b1[nt * 16 + nl];
        v8f acc;
        #pragma unroll
        for (int r = 0; r < 8; ++r) acc[r] = binit;
        #pragma unroll
        for (int kt = 0; kt < 5; ++kt) {
            // B frag: lane nl = column; 16 contiguous f16 of W1[n][kt*32+16g ..]
            v16h bfr = *(const v16h*)&s_w1[(nt * 16 + nl) * K1PAD + kt * 32 + 16 * g];
            acc = __builtin_amdgcn_wmma_f32_16x16x32_f16(false, a1[kt], false, bfr,
                                                         (short)0, acc, false, false);
        }
        #pragma unroll
        for (int r = 0; r < 8; ++r) {
            float v = acc[r] > 0.0f ? acc[r] : 0.0f;
            s_h1[(wave * 16 + r + 8 * g) * HIDDEN + nt * 16 + nl] = (_Float16)v;
        }
    }

    // --- GEMM2: h1(16x128) x W2^T, bias -> LDS atomic segment accumulation -
    v16h a2[4];
    #pragma unroll
    for (int kt = 0; kt < 4; ++kt) {
        const _Float16* rp = &s_h1[(wave * 16 + nl) * HIDDEN + kt * 32 + 16 * g];
        FragU u;
        u.h[0] = *(const v8h*)(rp);
        u.h[1] = *(const v8h*)(rp + 8);
        a2[kt] = u.v;
    }

    #pragma unroll
    for (int nt = 0; nt < 8; ++nt) {
        const float binit = b2[nt * 16 + nl];
        v8f acc;
        #pragma unroll
        for (int r = 0; r < 8; ++r) acc[r] = binit;
        #pragma unroll
        for (int kt = 0; kt < 4; ++kt) {
            v16h bfr = *(const v16h*)&s_w2[(nt * 16 + nl) * HIDDEN + kt * 32 + 16 * g];
            acc = __builtin_amdgcn_wmma_f32_16x16x32_f16(false, a2[kt], false, bfr,
                                                         (short)0, acc, false, false);
        }
        #pragma unroll
        for (int r = 0; r < 8; ++r) {
            int e_local = wave * 16 + r + 8 * g;       // edge within block
            atomicAdd(&s_acc[(e_local / KNN_K) * HIDDEN + nt * 16 + nl], acc[r]);
        }
    }
    __syncthreads();

    for (int i = tid; i < SPB * HIDDEN; i += LTHREADS) {
        int s = i >> 7;
        if (seg0 + s < T)
            x_out[(size_t)(seg0 + s) * HIDDEN + (i & (HIDDEN - 1))] = s_acc[i];
    }
}

// ---------------------------------------------------------------------------
// 6. Final classifier: sigmoid(feats @ Wf^T + bf)
// ---------------------------------------------------------------------------
__global__ void final_kernel(const float* __restrict__ x,
                             const float* __restrict__ wf,
                             const float* __restrict__ bfp,
                             float* __restrict__ out, int T)
{
    int t = blockIdx.x * blockDim.x + threadIdx.x;
    if (t >= T) return;
    float s = bfp[0];
    #pragma unroll 4
    for (int c = 0; c < HIDDEN; ++c) s += x[(size_t)t * HIDDEN + c] * wf[c];
    out[t] = 1.0f / (1.0f + expf(-s));
}

// ---------------------------------------------------------------------------
extern "C" void kernel_launch(void* const* d_in, const int* in_sizes, int n_in,
                              void* d_out, int out_size, void* d_ws, size_t ws_size,
                              hipStream_t stream)
{
    const float* points     = (const float*)d_in[0];
    const int*   triangles  = (const int*)d_in[1];
    const float* init_probs = (const float*)d_in[2];
    const float* W1[3] = {(const float*)d_in[3],  (const float*)d_in[7],  (const float*)d_in[11]};
    const float* B1[3] = {(const float*)d_in[4],  (const float*)d_in[8],  (const float*)d_in[12]};
    const float* W2[3] = {(const float*)d_in[5],  (const float*)d_in[9],  (const float*)d_in[13]};
    const float* B2[3] = {(const float*)d_in[6],  (const float*)d_in[10], (const float*)d_in[14]};
    const float* Wf = (const float*)d_in[15];
    const float* bf = (const float*)d_in[16];
    float* out = (float*)d_out;

    const int T = in_sizes[2];
    (void)n_in; (void)out_size; (void)ws_size;

    size_t off = 0;
    auto take = [&](size_t bytes) -> char* {
        char* p = (char*)d_ws + off;
        off += (bytes + 255) & ~(size_t)255;
        return p;
    };
    float* geo = (float*)take((size_t)T * 9 * sizeof(float));
    int*   nbr = (int*)take((size_t)T * KNN_K * sizeof(int));
    float* xA  = (float*)take((size_t)T * HIDDEN * sizeof(float));
    float* xB  = (float*)take((size_t)T * HIDDEN * sizeof(float));
    _Float16* wc[3];                      // fused W1|W2 fp16 image per layer
    for (int l = 0; l < 3; ++l)
        wc[l] = (_Float16*)take((size_t)SW_BYTES);

    // allow >64KB dynamic LDS for the fused layer kernel (320KB/WGP on CDNA5)
    (void)hipFuncSetAttribute((const void*)triconv_kernel,
                              hipFuncAttributeMaxDynamicSharedMemorySize,
                              SMEM_TOTAL);

    geometry_kernel<<<(T + 255) / 256, 256, 0, stream>>>(points, triangles, geo, T);
    knn_kernel<<<(T + 255) / 256, 256, 0, stream>>>(geo, nbr, T);
    for (int l = 0; l < 3; ++l) {
        convert_pad_kernel<<<(W1HALFS + 255) / 256, 256, 0, stream>>>(
            W1[l], wc[l], HIDDEN, HIDDEN + 9, K1PAD);
        convert_pad_kernel<<<(W2HALFS + 255) / 256, 256, 0, stream>>>(
            W2[l], wc[l] + W1HALFS, HIDDEN, HIDDEN, HIDDEN);
    }
    init_feats_kernel<<<((unsigned)(T * HIDDEN) + 255) / 256, 256, 0, stream>>>(init_probs, xA, T);

    const int lblocks = (T + SPB - 1) / SPB;
    triconv_kernel<<<lblocks, LTHREADS, SMEM_TOTAL, stream>>>(geo, nbr, xA, wc[0], B1[0], B2[0], xB, T);
    triconv_kernel<<<lblocks, LTHREADS, SMEM_TOTAL, stream>>>(geo, nbr, xB, wc[1], B1[1], B2[1], xA, T);
    triconv_kernel<<<lblocks, LTHREADS, SMEM_TOTAL, stream>>>(geo, nbr, xA, wc[2], B1[2], B2[2], xB, T);

    final_kernel<<<(T + 255) / 256, 256, 0, stream>>>(xB, Wf, bf, out, T);
}
